// MultiLayerGCN_10161892622615
// MI455X (gfx1250) — compile-verified, matrix operand order checked by reading, not run on hardware
//
#include <hip/hip_runtime.h>

// CDNA5 (gfx1250) wave32 WMMA types
typedef __attribute__((ext_vector_type(16))) __bf16 v16bf;
typedef __attribute__((ext_vector_type(8)))  __bf16 v8bf;
typedef __attribute__((ext_vector_type(8)))  float  v8f;

#define HC   128   // hidden channels
#define OUTC 64    // output channels

// ---------------- degree / normalization ----------------
__global__ void k_fill1(float* p, int n) {
    int i = blockIdx.x * blockDim.x + threadIdx.x;
    if (i < n) p[i] = 1.0f;                 // self-loop contributes 1 to deg
}

__global__ void k_deg(float* deg, const int* __restrict__ ei, int E) {
    int e = blockIdx.x * blockDim.x + threadIdx.x;
    if (e < E) atomicAdd(&deg[ei[E + e]], 1.0f);   // dst = ei[1][e]
}

__global__ void k_rsqrt(float* p, int n) {
    int i = blockIdx.x * blockDim.x + threadIdx.x;
    if (i < n) p[i] = rsqrtf(p[i]);          // deg >= 1 always (self loops)
}

// ---------------- layer 1 (rank-1: scalar propagation) ----------------
__global__ void k_l1_init(const float* __restrict__ x, const float* __restrict__ dinv,
                          float* s, float* t, int n) {
    int i = blockIdx.x * blockDim.x + threadIdx.x;
    if (i < n) { float v = x[i] * dinv[i]; s[i] = v; t[i] = v; }  // t init = self loop
}

__global__ void k_scat_scalar(const float* __restrict__ s, float* t,
                              const int* __restrict__ ei, int E) {
    int e = blockIdx.x * blockDim.x + threadIdx.x;
    if (e < E) atomicAdd(&t[ei[E + e]], s[ei[e]]);
}

// expand to bf16 activations (GEMM input precision anyway)
__global__ void k_l1_expand(const float* __restrict__ t, const float* __restrict__ dinv,
                            const float* __restrict__ W1, const float* __restrict__ b1,
                            __bf16* __restrict__ out, int n) {
    long tid = (long)blockIdx.x * blockDim.x + threadIdx.x;
    if (tid >= (long)n * HC) return;
    int node = (int)(tid >> 7), c = (int)(tid & (HC - 1));
    float v = t[node] * dinv[node] * W1[c] + b1[c];
    out[tid] = (__bf16)fmaxf(v, 0.0f);       // relu
}

// ---------------- vector propagation (layers 2,3) ----------------
// one wave per edge; lane handles a float4 slice of the 128-wide row.
// g and acc (51 MB each) are L2-resident on MI455X (192 MB L2): default RT hints.
__global__ void k_scat_vec(const float* __restrict__ g, float* acc,
                           const int* __restrict__ ei, int E) {
    int e = blockIdx.x * 8 + (threadIdx.x >> 5);
    if (e >= E) return;
    int lane = threadIdx.x & 31;
    int src = ei[e], dst = ei[E + e];
    const float4 v = *(const float4*)(g + (size_t)src * HC + lane * 4);
    float* p = acc + (size_t)dst * HC + lane * 4;
    atomicAdd(p + 0, v.x); atomicAdd(p + 1, v.y);
    atomicAdd(p + 2, v.z); atomicAdd(p + 3, v.w);
}

__global__ void k_finalize(const float* __restrict__ acc, const float* __restrict__ dinv,
                           const float* __restrict__ b, __bf16* __restrict__ out,
                           int n, int relu) {
    long tid = (long)blockIdx.x * blockDim.x + threadIdx.x;
    if (tid >= (long)n * HC) return;
    int node = (int)(tid >> 7), c = (int)(tid & (HC - 1));
    float v = acc[tid] * dinv[node] + b[c];
    if (relu) v = fmaxf(v, 0.0f);
    out[tid] = (__bf16)v;
}

// ---------------- weight prepack: f32 W[K,M] -> bf16 fragment order ----------------
// Wp[((coltile*nchunk + chunk)*32 + lane)*16 + j], j=2v+p per ISA 7.12.2 B layout,
// so a wave's B fragment is one contiguous 32-byte load.
__global__ void k_prepack_w(const float* __restrict__ W, __bf16* __restrict__ Wp,
                            int K, int M) {
    int tid = blockIdx.x * blockDim.x + threadIdx.x;
    if (tid >= K * M) return;
    int j  = tid & 15;
    int l  = (tid >> 4) & 31;
    int rest = tid >> 9;
    int nchunk = K >> 5;
    int chunk  = rest % nchunk;
    int ct     = rest / nchunk;
    int v = j >> 1, p = j & 1;
    int hf = l >> 4, m = l & 15;
    int kk  = (chunk << 5) + ((v < 4) ? (2 * v + 8 * hf) : (16 + 2 * (v - 4) + 8 * hf)) + p;
    int col = (ct << 4) + m;
    Wp[tid] = (__bf16)W[(size_t)kk * M + col];
}

// ---------------- WMMA GEMM ----------------
// C[N,M] = act((A[N,K]bf16 @ Wp) * rowscale[n] + bias[c]), fp32 accumulate/output.
// blockDim.x = (M/16)*32 (one wave per 16-col tile), grid.x = ceil(N/64):
// each wave computes 4 row-subtiles (64 rows x 16 cols) reusing the B fragment.
// A fragment = two b128 bf16 loads; B fragment = one contiguous 32B load.
template<int K, bool ROWSCALE, bool BIAS, bool RELU>
__global__ void gemm_wmma_bf16(const __bf16* __restrict__ A, const __bf16* __restrict__ Wp,
                               const float* __restrict__ rowscale, const float* __restrict__ bias,
                               float* __restrict__ C, int N, int M) {
    const int lane = threadIdx.x & 31;
    const int wave = threadIdx.x >> 5;
    const int row0 = blockIdx.x * 64;
    const int m    = lane & 15;
    const int hf   = lane >> 4;
    constexpr int NCH = K / 32;

    const __bf16* arowp[4];
#pragma unroll
    for (int rt = 0; rt < 4; ++rt) {
        int r = row0 + rt * 16 + m;
        if (r >= N) r = N - 1;               // clamp pad-row reads
        arowp[rt] = A + (size_t)r * K;
    }
    const __bf16* wp = Wp + ((size_t)wave * NCH * 32 + lane) * 16;

    v8f acc[4] = {v8f{}, v8f{}, v8f{}, v8f{}};
#pragma unroll
    for (int c = 0; c < NCH; ++c) {
        const v16bf bfrag = *(const v16bf*)(wp + (size_t)c * 32 * 16);
        const int kc = c * 32;
#pragma unroll
        for (int rt = 0; rt < 4; ++rt) {
            const v8bf lo = *(const v8bf*)(arowp[rt] + kc + 8 * hf);
            const v8bf hi = *(const v8bf*)(arowp[rt] + kc + 16 + 8 * hf);
            const v16bf afrag = __builtin_shufflevector(
                lo, hi, 0, 1, 2, 3, 4, 5, 6, 7, 8, 9, 10, 11, 12, 13, 14, 15);
            acc[rt] = __builtin_amdgcn_wmma_f32_16x16x32_bf16(
                false, afrag, false, bfrag, (short)0, acc[rt], false, false);
        }
    }

    // epilogue: compile-time variant, wave-uniform bounds fast path
    const int ncol = wave * 16 + m;
    const float bv = BIAS ? bias[ncol] : 0.0f;
#pragma unroll
    for (int rt = 0; rt < 4; ++rt) {
        const int rbase = row0 + rt * 16 + 8 * hf;   // C/D: VGPR r -> M = r + 8*half
        if (row0 + rt * 16 + 15 < N) {               // wave-uniform fast path
#pragma unroll
            for (int r = 0; r < 8; ++r) {
                const int rr = rbase + r;
                float v = acc[rt][r];
                if (ROWSCALE) v *= rowscale[rr];
                if (BIAS)     v += bv;
                if (RELU)     v = fmaxf(v, 0.0f);
                C[(size_t)rr * M + ncol] = v;
            }
        } else {
#pragma unroll
            for (int r = 0; r < 8; ++r) {
                const int rr = rbase + r;
                if (rr < N) {
                    float v = acc[rt][r];
                    if (ROWSCALE) v *= rowscale[rr];
                    if (BIAS)     v += bv;
                    if (RELU)     v = fmaxf(v, 0.0f);
                    C[(size_t)rr * M + ncol] = v;
                }
            }
        }
    }
}

extern "C" void kernel_launch(void* const* d_in, const int* in_sizes, int n_in,
                              void* d_out, int out_size, void* d_ws, size_t ws_size,
                              hipStream_t stream) {
    const float* x  = (const float*)d_in[0];
    const int*   ei = (const int*)  d_in[1];
    const float* W1 = (const float*)d_in[2];
    const float* b1 = (const float*)d_in[3];
    const float* W2 = (const float*)d_in[4];
    const float* b2 = (const float*)d_in[5];
    const float* W3 = (const float*)d_in[6];
    const float* b3 = (const float*)d_in[7];
    const float* Wo = (const float*)d_in[8];
    const float* bo = (const float*)d_in[9];

    const int N = in_sizes[0];          // x is [N,1]
    const int E = in_sizes[1] / 2;      // edge_index is [2,E]

    // workspace layout: dinv | s | t | g | acc  (f32)  then xbuf | W2p | W3p | Wop (bf16)
    float* ws     = (float*)d_ws;
    float*  dinv  = ws;
    float*  s     = ws + (size_t)N;
    float*  t     = ws + 2 * (size_t)N;
    float*  g     = ws + 3 * (size_t)N;
    float*  accb  = g    + (size_t)N * HC;
    __bf16* xbuf  = (__bf16*)(accb + (size_t)N * HC);
    __bf16* W2p   = xbuf + (size_t)N * HC;
    __bf16* W3p   = W2p + (size_t)HC * HC;
    __bf16* Wop   = W3p + (size_t)HC * HC;
    const size_t rowBytes = (size_t)N * HC * sizeof(float);

    const int gN   = (N + 255) / 256;
    const int gNH  = (int)(((long)N * HC + 255) / 256);
    const int gE   = (E + 255) / 256;
    const int gEv  = (E + 7) / 8;           // 8 waves (edges) per 256-thread block
    const int gR64 = (N + 63) / 64;         // 64 rows per block
    const int gWp  = (HC * HC + 255) / 256;
    const int gWpo = (HC * OUTC + 255) / 256;

    // weight prepack (tiny, once per call)
    k_prepack_w <<<gWp,  256, 0, stream>>>(W2, W2p, HC, HC);
    k_prepack_w <<<gWp,  256, 0, stream>>>(W3, W3p, HC, HC);
    k_prepack_w <<<gWpo, 256, 0, stream>>>(Wo, Wop, HC, OUTC);

    // symmetric normalization: dinv = rsqrt(deg), deg includes self loop
    k_fill1 <<<gN, 256, 0, stream>>>(dinv, N);
    k_deg   <<<gE, 256, 0, stream>>>(dinv, ei, E);
    k_rsqrt <<<gN, 256, 0, stream>>>(dinv, N);

    // ---- layer 1: rank-1 -> scalar propagation, then expand with W1/b1 + relu
    k_l1_init     <<<gN, 256, 0, stream>>>(x, dinv, s, t, N);
    k_scat_scalar <<<gE, 256, 0, stream>>>(s, t, ei, E);
    k_l1_expand   <<<gNH, 256, 0, stream>>>(t, dinv, W1, b1, xbuf, N);

    // ---- layer 2: g = (xbuf @ W2)*dinv ; acc = g (self loop) + scatter ; relu finalize
    gemm_wmma_bf16<HC, true, false, false>
        <<<gR64, (HC / 16) * 32, 0, stream>>>(xbuf, W2p, dinv, nullptr, g, N, HC);
    hipMemcpyAsync(accb, g, rowBytes, hipMemcpyDeviceToDevice, stream);
    k_scat_vec <<<gEv, 256, 0, stream>>>(g, accb, ei, E);
    k_finalize <<<gNH, 256, 0, stream>>>(accb, dinv, b2, xbuf, N, 1);

    // ---- layer 3: same, no relu
    gemm_wmma_bf16<HC, true, false, false>
        <<<gR64, (HC / 16) * 32, 0, stream>>>(xbuf, W3p, dinv, nullptr, g, N, HC);
    hipMemcpyAsync(accb, g, rowBytes, hipMemcpyDeviceToDevice, stream);
    k_scat_vec <<<gEv, 256, 0, stream>>>(g, accb, ei, E);
    k_finalize <<<gNH, 256, 0, stream>>>(accb, dinv, b3, xbuf, N, 0);

    // ---- head: out = xbuf @ Wo + bo   (N x 128 @ 128 x 64)
    gemm_wmma_bf16<HC, false, true, false>
        <<<gR64, (OUTC / 16) * 32, 0, stream>>>(xbuf, Wop, nullptr, bo,
                                                (float*)d_out, N, OUTC);
}
